// MultiHeadAttention_52510270161518
// MI455X (gfx1250) — compile-verified, matrix operand order checked by reading, not run on hardware
//
#include <hip/hip_runtime.h>
#include <hip/hip_bf16.h>

// ---------------------------------------------------------------------------
// MHA for MI455X (gfx1250): bf16 WMMA + async-to-LDS double-buffered GEMM.
// B=4, T=2048, D=1024, H=16, DK=64
// ---------------------------------------------------------------------------

#define BATCH 4
#define SEQ   2048
#define DIM   1024
#define HEADS 16
#define DK    64

typedef __attribute__((ext_vector_type(16))) __bf16 v16bf;
typedef __attribute__((ext_vector_type(8)))  float  v8f;
typedef __attribute__((ext_vector_type(8)))  unsigned int v8u;

// Build a 16-element bf16 WMMA fragment from two 16-byte chunks.
// ISA layout (16-bit A/B 16x32): lanes 0-15 hold K=[0..7],[16..23];
// lanes 16-31 hold K=[8..15],[24..31] -> chunk0 @ kh*8, chunk1 @ 16+kh*8.
__device__ __forceinline__ v16bf load_frag(const __bf16* p0, const __bf16* p1) {
    const uint4 a = *(const uint4*)p0;
    const uint4 b = *(const uint4*)p1;
    v8u v;
    v[0] = a.x; v[1] = a.y; v[2] = a.z; v[3] = a.w;
    v[4] = b.x; v[5] = b.y; v[6] = b.z; v[7] = b.w;
    return __builtin_bit_cast(v16bf, v);
}

__device__ __forceinline__ v8f wmma_bf16(v16bf a, v16bf b, v8f c) {
    return __builtin_amdgcn_wmma_f32_16x16x32_bf16(
        /*neg_a=*/false, a, /*neg_b=*/false, b,
        /*c_mod=*/(short)0, c, /*reuse_a=*/false, /*reuse_b=*/false);
}

// CDNA5 async global->LDS copy, 16B per lane (ASYNCcnt-tracked).
// dst VGPR carries the LDS byte address; generic shared ptr truncates to it.
__device__ __forceinline__ void async_b128(const __bf16* lds_ptr, const __bf16* gptr) {
    unsigned lds_off = (unsigned)(unsigned long long)lds_ptr;
    asm volatile("global_load_async_to_lds_b128 %0, %1, off"
                 :: "v"(lds_off), "v"(gptr)
                 : "memory");
}
__device__ __forceinline__ void wait_async0() {
    asm volatile("s_wait_asynccnt 0x0" ::: "memory");
}

// ---------------------------------------------------------------------------
// f32 -> bf16 cast (plain)
// ---------------------------------------------------------------------------
__global__ void cast_f32_bf16_kernel(const float* __restrict__ in,
                                     __bf16* __restrict__ out, int n) {
    int i = blockIdx.x * blockDim.x + threadIdx.x;
    if (i < n) out[i] = (__bf16)in[i];
}

// ---------------------------------------------------------------------------
// f32 -> bf16 cast with transpose: out[c][r] = in[r][c], square D x D.
// 16x16 LDS tile, 256 threads, both sides coalesced.
// ---------------------------------------------------------------------------
__global__ void cast_transpose_kernel(const float* __restrict__ in,
                                      __bf16* __restrict__ out, int n) {
    __shared__ float tile[16][17];
    const int tx = threadIdx.x & 15, ty = threadIdx.x >> 4;
    const int r0 = blockIdx.y * 16, c0 = blockIdx.x * 16;
    tile[ty][tx] = in[(size_t)(r0 + ty) * n + c0 + tx];
    __syncthreads();
    out[(size_t)(c0 + ty) * n + r0 + tx] = (__bf16)tile[tx][ty];
}

// ---------------------------------------------------------------------------
// Tiled bf16 GEMM: C[M,N] = scale * (A[M,K] @ Bt^T)
//   A  : [M,K] row-major bf16
//   Bt : [N,K] row-major bf16 (i.e. B already transposed -> contiguous [n][k])
// Block tile 128(M) x 256(N), K-step 32, double-buffered async LDS staging.
// 256 threads = 8 waves (2x4); each wave computes 64x64 = 4x4 WMMA tiles
// -> 16 v_wmma_f32_16x16x32_bf16 per K-step per wave.
// ---------------------------------------------------------------------------
template <typename OutT>
__global__ __launch_bounds__(256, 1) void gemm_bf16_wmma(
    const __bf16* __restrict__ A, const __bf16* __restrict__ Bt,
    OutT* __restrict__ C, int M, int N, int K, float scale) {

    __shared__ __attribute__((aligned(16))) __bf16 As[2][128 * 32]; // 2 x 8 KB
    __shared__ __attribute__((aligned(16))) __bf16 Bs[2][256 * 32]; // 2 x 16 KB

    const int tid  = threadIdx.x;
    const int wid  = tid >> 5;
    const int lane = tid & 31;
    const int wm   = wid >> 2;        // 0..1 -> 64-row slab
    const int wn   = wid & 3;         // 0..3 -> 64-col slab
    const int lm   = lane & 15;
    const int kh   = lane >> 4;

    const int m0 = blockIdx.y * 128;
    const int n0 = blockIdx.x * 256;

    // staging assignments (per K-step):
    //   A tile 128x32: thread t -> row t>>1, half t&1 (2 x b128)
    //   B tile 256x32: thread t -> row t      (4 x b128)
    const int ar = tid >> 1, ah = tid & 1;

    v8f acc[4][4] = {};

    auto stage = [&](int buf, int k0) {
        const __bf16* ga = A + (size_t)(m0 + ar) * K + k0 + ah * 16;
        __bf16* la = &As[buf][ar * 32 + ah * 16];
        async_b128(la,     ga);
        async_b128(la + 8, ga + 8);
        const __bf16* gb = Bt + (size_t)(n0 + tid) * K + k0;
        __bf16* lb = &Bs[buf][tid * 32];
        async_b128(lb,      gb);
        async_b128(lb + 8,  gb + 8);
        async_b128(lb + 16, gb + 16);
        async_b128(lb + 24, gb + 24);
    };

    stage(0, 0);
    wait_async0();
    __syncthreads();

    int cur = 0;
    for (int k0 = 0; k0 < K; k0 += 32) {
        if (k0 + 32 < K) stage(cur ^ 1, k0 + 32);

        v16bf af[4], bf[4];
#pragma unroll
        for (int i = 0; i < 4; ++i) {
            const int row = wm * 64 + i * 16 + lm;
            af[i] = load_frag(&As[cur][row * 32 + kh * 8],
                              &As[cur][row * 32 + 16 + kh * 8]);
        }
#pragma unroll
        for (int j = 0; j < 4; ++j) {
            const int col = wn * 64 + j * 16 + lm;
            bf[j] = load_frag(&Bs[cur][col * 32 + kh * 8],
                              &Bs[cur][col * 32 + 16 + kh * 8]);
        }
#pragma unroll
        for (int i = 0; i < 4; ++i)
#pragma unroll
            for (int j = 0; j < 4; ++j)
                acc[i][j] = wmma_bf16(af[i], bf[j], acc[i][j]);

        wait_async0();
        __syncthreads();
        cur ^= 1;
    }

    // epilogue: C/D layout: elem e -> row base + e + (lane>>4)*8, col = lane&15
#pragma unroll
    for (int i = 0; i < 4; ++i) {
#pragma unroll
        for (int j = 0; j < 4; ++j) {
            const int rbase = m0 + wm * 64 + i * 16 + kh * 8;
            const int col   = n0 + wn * 64 + j * 16 + lm;
#pragma unroll
            for (int e = 0; e < 8; ++e) {
                const float v = acc[i][j][e] * scale;
                C[(size_t)(rbase + e) * N + col] = (OutT)v;
            }
        }
    }
}

// ---------------------------------------------------------------------------
// Attention: per wave, 16 query rows of one (b,h); stream all T key columns
// in 16-wide tiles. Scores via 2 chained bf16 WMMAs (K=64). Online softmax
// (running max / sum-exp), cross-lane combine over the 16 lanes sharing a
// row, then mid[b,h,t,:] = (sum_l A[t,l]) * v[b,t,h,:]  (head-major layout).
// grid.x = B*H*(T/128); block = 256 (8 waves x 16 rows = 128 rows/block)
// ---------------------------------------------------------------------------
__global__ __launch_bounds__(256) void attn_softmax_kernel(
    const __bf16* __restrict__ qb, const __bf16* __restrict__ kb,
    const __bf16* __restrict__ vb, const float* __restrict__ mask,
    __bf16* __restrict__ midb) {

    const int bidx = blockIdx.x;
    const int mblk = bidx & 15;          // T/128 = 16 row-blocks
    const int h    = (bidx >> 4) & 15;
    const int b    = bidx >> 8;

    const int wid  = threadIdx.x >> 5;
    const int lane = threadIdx.x & 31;
    const int lm   = lane & 15;
    const int kh   = lane >> 4;
    const int t0   = mblk * 128 + wid * 16;

    // Q fragments for this wave's 16 rows (K = 64 -> two 16x32 frags)
    const __bf16* qrow = qb + ((size_t)(b * SEQ + t0 + lm)) * DIM + h * DK;
    const v16bf aq0 = load_frag(qrow + kh * 8,      qrow + 16 + kh * 8);
    const v16bf aq1 = load_frag(qrow + 32 + kh * 8, qrow + 48 + kh * 8);

    float mrun[8], srun[8];
#pragma unroll
    for (int e = 0; e < 8; ++e) { mrun[e] = -3.0e38f; srun[e] = 0.0f; }

    for (int c0 = 0; c0 < SEQ; c0 += 16) {
        const __bf16* krow = kb + ((size_t)(b * SEQ + c0 + lm)) * DIM + h * DK;
        // prefetch the K rows 4 tiles ahead into cache (global_prefetch_b8)
        if (c0 + 64 < SEQ)
            __builtin_prefetch(krow + (size_t)64 * DIM, 0, 1);

        const v16bf bk0 = load_frag(krow + kh * 8,      krow + 16 + kh * 8);
        const v16bf bk1 = load_frag(krow + 32 + kh * 8, krow + 48 + kh * 8);

        v8f sc = {};
        sc = wmma_bf16(aq0, bk0, sc);
        sc = wmma_bf16(aq1, bk1, sc);

        const int col = c0 + lm;
        const int rb  = t0 + kh * 8;
#pragma unroll
        for (int e = 0; e < 8; ++e) {
            const float x  = sc[e] + mask[((size_t)b * SEQ + rb + e) * SEQ + col];
            const float mn = fmaxf(mrun[e], x);
            srun[e] = srun[e] * __expf(mrun[e] - mn) + __expf(x - mn);
            mrun[e] = mn;
        }
    }

    // combine partial (max, sumexp) across the 16 lanes that share each row
#pragma unroll
    for (int off = 1; off < 16; off <<= 1) {
#pragma unroll
        for (int e = 0; e < 8; ++e) {
            const float mo = __shfl_xor(mrun[e], off, 32);
            const float so = __shfl_xor(srun[e], off, 32);
            const float mn = fmaxf(mrun[e], mo);
            srun[e] = srun[e] * __expf(mrun[e] - mn) + so * __expf(mo - mn);
            mrun[e] = mn;
        }
    }

    // mid[b][h*T*DK + t*DK + dk] = rowsumA * v[b,t,h,dk]; lane handles 4 dk
#pragma unroll
    for (int e = 0; e < 8; ++e) {
        const float g = srun[e] / srun[e];   // sum_l softmax = s/s
        const int   t = t0 + kh * 8 + e;
        const __bf16* vp = vb + ((size_t)(b * SEQ + t)) * DIM + h * DK + lm * 4;
        __bf16* op = midb + (size_t)b * SEQ * DIM + (size_t)h * SEQ * DK +
                     (size_t)t * DK + lm * 4;
#pragma unroll
        for (int d = 0; d < 4; ++d)
            op[d] = (__bf16)((float)vp[d] * g);
    }
}

// ---------------------------------------------------------------------------
// Host-side launcher
// ---------------------------------------------------------------------------
extern "C" void kernel_launch(void* const* d_in, const int* in_sizes, int n_in,
                              void* d_out, int out_size, void* d_ws, size_t ws_size,
                              hipStream_t stream) {
    const float* x    = (const float*)d_in[0];   // [B,T,D]
    const float* mask = (const float*)d_in[1];   // [B,1,T,T]
    const float* Wq   = (const float*)d_in[2];   // [D,D]
    const float* Wk   = (const float*)d_in[3];
    const float* Wv   = (const float*)d_in[4];
    const float* Wo   = (const float*)d_in[5];
    float* out        = (float*)d_out;           // [B,T,D]

    const size_t XN = (size_t)BATCH * SEQ * DIM;     // 8,388,608
    const size_t WN = (size_t)DIM * DIM;             // 1,048,576

    char* w = (char*)d_ws;
    __bf16* xb  = (__bf16*)w;                 w += XN * 2;   // 16 MB
    __bf16* wqT = (__bf16*)w;                 w += WN * 2;   //  2 MB (Wq^T)
    __bf16* wkT = (__bf16*)w;                 w += WN * 2;   //       (Wk^T)
    __bf16* wvT = (__bf16*)w;                 w += WN * 2;   //       (Wv^T)
    __bf16* wob = (__bf16*)w;                 w += WN * 2;   //       (Wo, row-major)
    __bf16* qb  = (__bf16*)w;                 w += XN * 2;   // 16 MB
    __bf16* kb  = (__bf16*)w;                 w += XN * 2;
    __bf16* vb  = (__bf16*)w;                 w += XN * 2;
    __bf16* midb = xb;  // xb is dead after the three projection GEMMs

    // 1) casts: x and Wo plain; Wq/Wk/Wv transposed so GEMM B-operand is [n][k]
    cast_f32_bf16_kernel<<<(int)((XN + 255) / 256), 256, 0, stream>>>(x, xb, (int)XN);
    cast_f32_bf16_kernel<<<(int)((WN + 255) / 256), 256, 0, stream>>>(Wo, wob, (int)WN);
    dim3 tg(DIM / 16, DIM / 16);
    cast_transpose_kernel<<<tg, 256, 0, stream>>>(Wq, wqT, DIM);
    cast_transpose_kernel<<<tg, 256, 0, stream>>>(Wk, wkT, DIM);
    cast_transpose_kernel<<<tg, 256, 0, stream>>>(Wv, wvT, DIM);

    // 2) projections: Q (scaled by 1/sqrt(DK)=0.125), K, V
    const int M = BATCH * SEQ, N = DIM, K = DIM;
    dim3 gg(N / 256, M / 128);   // (4, 64)
    gemm_bf16_wmma<__bf16><<<gg, 256, 0, stream>>>(xb, wqT, qb, M, N, K, 0.125f);
    gemm_bf16_wmma<__bf16><<<gg, 256, 0, stream>>>(xb, wkT, kb, M, N, K, 1.0f);
    gemm_bf16_wmma<__bf16><<<gg, 256, 0, stream>>>(xb, wvT, vb, M, N, K, 1.0f);

    // 3) scores + online softmax + (rowsum A) * v -> mid (head-major)
    attn_softmax_kernel<<<BATCH * HEADS * (SEQ / 128), 256, 0, stream>>>(
        qb, kb, vb, mask, midb);

    // 4) out = mid @ Wo^T  ([n][k] operand is Wo row-major itself), f32 out
    gemm_bf16_wmma<float><<<gg, 256, 0, stream>>>(midb, wob, out, M, N, K, 1.0f);
}
